// ViTMHAttention_86157043958042
// MI455X (gfx1250) — matrix operand order repaired and checked by reading, not compiled, not run on hardware
//
#include <hip/hip_runtime.h>
#include <hip/hip_bf16.h>
#include <stdint.h>

#define B_  32
#define T_  577
#define E_  1024
#define H_  16
#define D_  64
#define TP  608              // T padded to multiple of 32
#define M_  (B_ * T_)        // 18464 rows

typedef __attribute__((ext_vector_type(16))) __bf16 v16bf;
typedef __attribute__((ext_vector_type(8)))  float  v8f;

union BF16Frag {
    v16bf        v;
    uint4        q[2];
    unsigned int u[8];
    unsigned short s[16];
};

__device__ __forceinline__ unsigned short f2bf(float f) {
    unsigned int u = __builtin_bit_cast(unsigned int, f);
    u += 0x7fffu + ((u >> 16) & 1u);          // round-to-nearest-even
    return (unsigned short)(u >> 16);
}

// ---------------------------------------------------------------- convert x
__global__ __launch_bounds__(256) void cvt_x_bf16(const float* __restrict__ in,
                                                  unsigned short* __restrict__ out,
                                                  int n8) {
    int i = blockIdx.x * blockDim.x + threadIdx.x;
    if (i >= n8) return;
    long base = (long)i * 8;
    float4 a = *(const float4*)(in + base);
    float4 b = *(const float4*)(in + base + 4);
    uint4 o;
    o.x = (unsigned)f2bf(a.x) | ((unsigned)f2bf(a.y) << 16);
    o.y = (unsigned)f2bf(a.z) | ((unsigned)f2bf(a.w) << 16);
    o.z = (unsigned)f2bf(b.x) | ((unsigned)f2bf(b.y) << 16);
    o.w = (unsigned)f2bf(b.z) | ((unsigned)f2bf(b.w) << 16);
    *(uint4*)(out + base) = o;
}

// ------------------------------------------- convert + transpose weight (LDS)
__global__ __launch_bounds__(256) void cvt_w_T(const float* __restrict__ w,
                                               unsigned short* __restrict__ wt) {
    __shared__ float tile[32][33];
    int bx = blockIdx.x * 32;   // k block (input row)
    int by = blockIdx.y * 32;   // n block (input col)
    int tx = threadIdx.x, ty = threadIdx.y;   // 32 x 8
#pragma unroll
    for (int i = 0; i < 32; i += 8)
        tile[ty + i][tx] = w[(long)(bx + ty + i) * E_ + (by + tx)];
    __syncthreads();
#pragma unroll
    for (int i = 0; i < 32; i += 8)
        wt[(long)(by + ty + i) * E_ + (bx + tx)] = f2bf(tile[tx][ty + i]);
}

// ---------------------------------------------------------------- WMMA GEMM
// C[M,N] = A[M,K=1024](bf16) @ Wt[N,K](bf16, row-major over K) + bias
// Wave tile 64x64 (4 M-tiles x 4 N-tiles) -> 16 wmma per 16 b128 loads/k-step.
// Block: 8 waves (2 M x 4 N) -> 128 x 256 block tile.
// mode 0: write Q -> [B,H,TP,D]   mode 1: write K -> [B,H,TP,D]
// mode 2: write V^T -> [B,H,D,TP] mode 3: write f32 -> out[M,E]
__global__ __launch_bounds__(256) void gemm16(
    const unsigned short* __restrict__ A,
    const unsigned short* __restrict__ WtQ, const unsigned short* __restrict__ WtK,
    const unsigned short* __restrict__ WtV,
    const float* __restrict__ bQ, const float* __restrict__ bK, const float* __restrict__ bV,
    unsigned short* __restrict__ dq, unsigned short* __restrict__ dk,
    unsigned short* __restrict__ dv, float* __restrict__ dO, int modeBase) {
    const int mode = modeBase + blockIdx.z;
    const unsigned short* Wt = (mode == 1) ? WtK : (mode == 2) ? WtV : WtQ;
    const float* bias        = (mode == 1) ? bK  : (mode == 2) ? bV  : bQ;

    const int lane = threadIdx.x & 31, wave = threadIdx.x >> 5;
    const int h = lane >> 4, ln = lane & 15;
    const int wm = wave >> 2, wn = wave & 3;
    const long m0 = (long)blockIdx.x * 128 + wm * 64;
    const long n0 = (long)blockIdx.y * 256 + wn * 64;

    v8f acc[4][4];
#pragma unroll
    for (int mt = 0; mt < 4; ++mt)
#pragma unroll
        for (int nt = 0; nt < 4; ++nt)
#pragma unroll
            for (int jj = 0; jj < 8; ++jj) acc[mt][nt][jj] = 0.f;

    const unsigned short* aptr[4];
#pragma unroll
    for (int mt = 0; mt < 4; ++mt) {
        long r = m0 + mt * 16 + ln;
        if (r >= M_) r = M_ - 1;              // clamp: guarded at store
        aptr[mt] = A + r * E_;
    }
    const unsigned short* bptr[4];
#pragma unroll
    for (int nt = 0; nt < 4; ++nt)
        bptr[nt] = Wt + (n0 + nt * 16 + ln) * E_ + h * 16;

    for (int k0 = 0; k0 < E_; k0 += 32) {
        BF16Frag a[4], b[4];
#pragma unroll
        for (int mt = 0; mt < 4; ++mt) {       // A 16x32: 32B contiguous at k0+h*16
            a[mt].q[0] = *(const uint4*)(aptr[mt] + k0 + h * 16);
            a[mt].q[1] = *(const uint4*)(aptr[mt] + k0 + h * 16 + 8);
        }
#pragma unroll
        for (int nt = 0; nt < 4; ++nt) {       // B 32x16: 32B contiguous
            b[nt].q[0] = *(const uint4*)(bptr[nt] + k0);
            b[nt].q[1] = *(const uint4*)(bptr[nt] + k0 + 8);
        }
#pragma unroll
        for (int mt = 0; mt < 4; ++mt)
#pragma unroll
            for (int nt = 0; nt < 4; ++nt)
                acc[mt][nt] = __builtin_amdgcn_wmma_f32_16x16x32_bf16(
                    false, a[mt].v, false, b[nt].v, (short)0, acc[mt][nt], false, false);
    }

#pragma unroll
    for (int nt = 0; nt < 4; ++nt) {
        const int n  = (int)(n0 + nt * 16 + ln);
        const float bn = bias[n];
        const int hd = n >> 6, d = n & 63;
#pragma unroll
        for (int mt = 0; mt < 4; ++mt)
#pragma unroll
            for (int jj = 0; jj < 8; ++jj) {
                long m = m0 + mt * 16 + jj + h * 8;
                if (m >= M_) continue;
                float val = acc[mt][nt][jj] + bn;
                if (mode == 3) {
                    dO[m * E_ + n] = val;
                } else {
                    int bb = (int)(m / T_), t = (int)(m - (long)bb * T_);
                    unsigned short bf = f2bf(val);
                    if (mode == 2)
                        dv[((long)(bb * H_ + hd) * D_ + d) * TP + t] = bf;     // V^T
                    else
                        ((mode == 0) ? dq : dk)[((long)(bb * H_ + hd) * TP + t) * D_ + d] = bf;
                }
            }
    }
}

// -------------------------------------------------------- flash attention
// One wave per 16-query tile; computes S^T = K @ Q^T so the C-layout of
// exp(S^T) matches the A-fragment layout of P for the P@V WMMA (no transpose).
__global__ __launch_bounds__(64) void attn(
    const unsigned short* __restrict__ qb,   // [B,H,TP,D]
    const unsigned short* __restrict__ kb,   // [B,H,TP,D]
    const unsigned short* __restrict__ vb,   // [B,H,D,TP]  (transposed)
    unsigned short* __restrict__ ctx) {      // [B,T,E]
    const int lane = threadIdx.x & 31, wave = threadIdx.x >> 5;
    const int h = lane >> 4, ln = lane & 15;
    const int qt = blockIdx.x * 2 + wave;
    const int head = blockIdx.y, bb = blockIdx.z;
    const int q0 = qt * 16;
    if (q0 >= T_) return;                    // wave-uniform, no barriers used

    const long bh = (long)(bb * H_ + head);
    const unsigned short* qp = qb + bh * TP * D_;
    const unsigned short* kp = kb + bh * TP * D_;
    const unsigned short* vp = vb + bh * (long)D_ * TP;

    // Q^T B-fragments, d-steps 0 (d 0..31) and 1 (d 32..63)
    BF16Frag bq0, bq1;
    {
        const unsigned short* p = qp + (long)(q0 + ln) * D_ + h * 8;
        bq0.q[0] = *(const uint4*)(p);      bq0.q[1] = *(const uint4*)(p + 16);
        bq1.q[0] = *(const uint4*)(p + 32); bq1.q[1] = *(const uint4*)(p + 48);
    }

    const float SCALE = 0.125f * 1.44269504089f;   // 1/sqrt(64) folded into exp2
    float mrun = -1e30f, lrun = 0.f;
    v8f o[4];
#pragma unroll
    for (int dt = 0; dt < 4; ++dt)
#pragma unroll
        for (int jj = 0; jj < 8; ++jj) o[dt][jj] = 0.f;

    for (int j = 0; j < TP; j += 32) {
        // K A-fragments: tile t2 covers keys j + t2*16 .. +15
        v8f st[2];
#pragma unroll
        for (int t2 = 0; t2 < 2; ++t2) {
            const unsigned short* p = kp + (long)(j + t2 * 16 + ln) * D_ + h * 8;
            BF16Frag ka0, ka1;
            ka0.q[0] = *(const uint4*)(p);       ka0.q[1] = *(const uint4*)(p + 16);
            ka1.q[0] = *(const uint4*)(p + 32);  ka1.q[1] = *(const uint4*)(p + 48);
            v8f z;
#pragma unroll
            for (int jj = 0; jj < 8; ++jj) z[jj] = 0.f;
            z = __builtin_amdgcn_wmma_f32_16x16x32_bf16(false, ka0.v, false, bq0.v,
                                                        (short)0, z, false, false);
            st[t2] = __builtin_amdgcn_wmma_f32_16x16x32_bf16(false, ka1.v, false, bq1.v,
                                                             (short)0, z, false, false);
        }
#pragma unroll
        for (int t2 = 0; t2 < 2; ++t2)
#pragma unroll
            for (int jj = 0; jj < 8; ++jj) st[t2][jj] *= SCALE;
        if (j + 32 > T_) {                      // mask padded keys
#pragma unroll
            for (int t2 = 0; t2 < 2; ++t2)
#pragma unroll
                for (int jj = 0; jj < 8; ++jj) {
                    int key = j + t2 * 16 + jj + h * 8;
                    if (key >= T_) st[t2][jj] = -1e30f;
                }
        }
        // online softmax (stats per lane: query = ln, duplicated across halves)
        float tm = st[0][0];
#pragma unroll
        for (int t2 = 0; t2 < 2; ++t2)
#pragma unroll
            for (int jj = 0; jj < 8; ++jj) tm = fmaxf(tm, st[t2][jj]);
        tm = fmaxf(tm, __shfl_xor(tm, 16));
        float mnew  = fmaxf(mrun, tm);
        float alpha = exp2f(mrun - mnew);
        float ps[16], rs = 0.f;
#pragma unroll
        for (int t2 = 0; t2 < 2; ++t2)
#pragma unroll
            for (int jj = 0; jj < 8; ++jj) {
                float e = exp2f(st[t2][jj] - mnew);
                ps[t2 * 8 + jj] = e;
                rs += e;
            }
        rs += __shfl_xor(rs, 16);
        lrun = lrun * alpha + rs;
        mrun = mnew;

        // pack P A-fragment (layout identity: element i = ps[i])
        BF16Frag ap;
#pragma unroll
        for (int i = 0; i < 8; ++i)
            ap.u[i] = (unsigned)f2bf(ps[2 * i]) | ((unsigned)f2bf(ps[2 * i + 1]) << 16);

        // rescale O rows (row q = jj + h*8 lives in lane jj + h*8's stats)
        float aj[8];
#pragma unroll
        for (int jj = 0; jj < 8; ++jj) aj[jj] = __shfl(alpha, h * 8 + jj);
#pragma unroll
        for (int dt = 0; dt < 4; ++dt)
#pragma unroll
            for (int jj = 0; jj < 8; ++jj) o[dt][jj] *= aj[jj];

        // O += P @ V, 4 d-tiles; V^T rows give contiguous B-fragments
#pragma unroll
        for (int dt = 0; dt < 4; ++dt) {
            const unsigned short* p = vp + (long)(dt * 16 + ln) * TP + j + h * 8;
            BF16Frag bv;
            bv.q[0] = *(const uint4*)(p);
            bv.q[1] = *(const uint4*)(p + 16);
            o[dt] = __builtin_amdgcn_wmma_f32_16x16x32_bf16(false, ap.v, false, bv.v,
                                                            (short)0, o[dt], false, false);
        }
    }

    float inv = 1.f / lrun;
    float ij[8];
#pragma unroll
    for (int jj = 0; jj < 8; ++jj) ij[jj] = __shfl(inv, h * 8 + jj);
#pragma unroll
    for (int dt = 0; dt < 4; ++dt)
#pragma unroll
        for (int jj = 0; jj < 8; ++jj) {
            int t = q0 + jj + h * 8;
            if (t < T_)
                ctx[((long)(bb * T_ + t)) * E_ + head * 64 + dt * 16 + ln] =
                    f2bf(o[dt][jj] * ij[jj]);
        }
}

// ----------------------------------------------------------------- launcher
extern "C" void kernel_launch(void* const* d_in, const int* in_sizes, int n_in,
                              void* d_out, int out_size, void* d_ws, size_t ws_size,
                              hipStream_t stream) {
    const float* x  = (const float*)d_in[0];
    const float* Wq = (const float*)d_in[1];
    const float* bq = (const float*)d_in[2];
    const float* Wk = (const float*)d_in[3];
    const float* bk = (const float*)d_in[4];
    const float* Wv = (const float*)d_in[5];
    const float* bv = (const float*)d_in[6];
    const float* Wo = (const float*)d_in[7];
    const float* bo = (const float*)d_in[8];
    float* out = (float*)d_out;

    char* ws = (char*)d_ws;
    size_t off = 0;
    unsigned short* xb  = (unsigned short*)(ws + off); off += (size_t)M_ * E_ * 2;
    unsigned short* wqT = (unsigned short*)(ws + off); off += (size_t)E_ * E_ * 2;
    unsigned short* wkT = (unsigned short*)(ws + off); off += (size_t)E_ * E_ * 2;
    unsigned short* wvT = (unsigned short*)(ws + off); off += (size_t)E_ * E_ * 2;
    unsigned short* woT = (unsigned short*)(ws + off); off += (size_t)E_ * E_ * 2;
    unsigned short* qB  = (unsigned short*)(ws + off); off += (size_t)B_ * H_ * TP * D_ * 2;
    unsigned short* kB  = (unsigned short*)(ws + off); off += (size_t)B_ * H_ * TP * D_ * 2;
    unsigned short* vB  = (unsigned short*)(ws + off); off += (size_t)B_ * H_ * TP * D_ * 2;
    unsigned short* ctx = xb;   // xb dead after QKV GEMM; reuse for context

    // 1) precision conversion (x -> bf16, W -> bf16 transposed)
    int n8 = (M_ * E_) / 8;
    cvt_x_bf16<<<(n8 + 255) / 256, 256, 0, stream>>>(x, xb, n8);
    dim3 tb(32, 8), tg(E_ / 32, E_ / 32);
    cvt_w_T<<<tg, tb, 0, stream>>>(Wq, wqT);
    cvt_w_T<<<tg, tb, 0, stream>>>(Wk, wkT);
    cvt_w_T<<<tg, tb, 0, stream>>>(Wv, wvT);
    cvt_w_T<<<tg, tb, 0, stream>>>(Wo, woT);

    // 2) fused Q/K/V projection (grid.z selects head matrix + layout)
    dim3 gg((M_ + 127) / 128, E_ / 256, 3);
    gemm16<<<gg, 256, 0, stream>>>(xb, wqT, wkT, wvT, bq, bk, bv,
                                   qB, kB, vB, nullptr, 0);

    // 3) flash attention, one wave per 16-query tile, 2 waves / block
    attn<<<dim3((TP / 16 + 1) / 2, H_, B_), 64, 0, stream>>>(qB, kB, vB, ctx);

    // 4) output projection -> f32
    dim3 go((M_ + 127) / 128, E_ / 256, 1);
    gemm16<<<go, 256, 0, stream>>>(ctx, woT, woT, woT, bo, bo, bo,
                                   nullptr, nullptr, nullptr, out, 3);
}
